// RNAStructureGNN_14396730376431
// MI455X (gfx1250) — compile-verified, matrix operand order checked by reading, not run on hardware
//
#include <hip/hip_runtime.h>
#include <hip/hip_bf16.h>
#include <math.h>

typedef _Float16 v16h __attribute__((ext_vector_type(16)));
typedef _Float16 v8h  __attribute__((ext_vector_type(8)));
typedef float    v8f  __attribute__((ext_vector_type(8)));
typedef int      v4i_ __attribute__((ext_vector_type(4)));

#define DD 128
#define HH_ 4
#define CC 32
#define NEG_SLOPE 0.2f
#define LN_EPS 1e-5f

#if defined(__HIP_DEVICE_COMPILE__) && defined(__gfx1250__) && \
    __has_builtin(__builtin_amdgcn_global_load_async_to_lds_b128)
#define HAVE_ASYNC_LDS 1
#endif

#ifdef HAVE_ASYNC_LDS
__device__ __forceinline__ void async_copy_b128(const void* g, void* l) {
  // builtin signature (from compiler diagnostic): (v4i AS1*, v4i AS3*, Imm, Imm)
  __builtin_amdgcn_global_load_async_to_lds_b128(
      (__attribute__((address_space(1))) v4i_*)(unsigned long long)g,
      (__attribute__((address_space(3))) v4i_*)(unsigned)(unsigned long long)l,
      0, 0);
}
__device__ __forceinline__ void wait_async0() {
#if __has_builtin(__builtin_amdgcn_s_wait_asynccnt)
  __builtin_amdgcn_s_wait_asynccnt(0);
#else
  asm volatile("s_wait_asynccnt 0x0" ::: "memory");
#endif
}
#endif

// ------- node embed: h = relu(x @ W0 + b0); keep f32 + f16 mirror ----------
__global__ void node_embed_kernel(const float* __restrict__ x,
                                  const float* __restrict__ W0,
                                  const float* __restrict__ b0,
                                  float* __restrict__ h,
                                  _Float16* __restrict__ h16, int N) {
  int gid = blockIdx.x * blockDim.x + threadIdx.x;
  if (gid >= N * DD) return;
  int n = gid >> 7, d = gid & 127;
  const float* xr = x + (size_t)n * 9;
  float acc = b0[d];
#pragma unroll
  for (int k = 0; k < 9; ++k) acc = fmaf(xr[k], W0[k * DD + d], acc);
  acc = fmaxf(acc, 0.f);
  h[gid] = acc;
  h16[gid] = (_Float16)acc;
}

// ------------- WMMA GEMM: C[M,128] = A16[M,128] @ W[128,128] (+bias,relu) ---
// A is f16 in global (async-copied straight into LDS); W staged f32->f16
// transposed; f32 accumulation via v_wmma_f32_16x16x32_f16.
__global__ __launch_bounds__(256)
void gemm128_wmma_kernel(const _Float16* __restrict__ A16,
                         const float* __restrict__ W,
                         const float* __restrict__ bias,
                         float* __restrict__ Cout32,
                         _Float16* __restrict__ Cout16,
                         int M, int do_relu) {
  __shared__ _Float16 As[128][128];      // 32 KB
  __shared__ _Float16 Bs[128][128];      // 32 KB, Bs[n][k] = W[k][n]
  const int tid  = threadIdx.x;
  const int row0 = blockIdx.x * 128;

  // A tile: 128 rows x 256B = 2048 x b128 chunks.
#ifdef HAVE_ASYNC_LDS
#pragma unroll
  for (int i = 0; i < 8; ++i) {
    int idx = tid + i * 256;
    int r = idx >> 4, c8 = (idx & 15) << 3;
    int grow = row0 + r;
    if (grow < M)  // stale rows only feed unstored D rows -> no zero-fill needed
      async_copy_b128(A16 + (size_t)grow * DD + c8, &As[r][c8]);
  }
#else
#pragma unroll
  for (int i = 0; i < 8; ++i) {
    int idx = tid + i * 256;
    int r = idx >> 4, c8 = (idx & 15) << 3;
    int grow = row0 + r;
    v8h v = {};
    if (grow < M) v = *(const v8h*)(A16 + (size_t)grow * DD + c8);
    *(v8h*)&As[r][c8] = v;
  }
#endif
  // W tile transposed so B fragments are contiguous in LDS.
#pragma unroll
  for (int i = 0; i < 16; ++i) {
    int idx = tid + i * 256;
    int k = idx >> 5, c4 = (idx & 31) << 2;
    float4 v = *(const float4*)(W + (size_t)k * DD + c4);
    Bs[c4 + 0][k] = (_Float16)v.x; Bs[c4 + 1][k] = (_Float16)v.y;
    Bs[c4 + 2][k] = (_Float16)v.z; Bs[c4 + 3][k] = (_Float16)v.w;
  }
#ifdef HAVE_ASYNC_LDS
  wait_async0();
#endif
  __syncthreads();

  const int wave = tid >> 5;             // 8 waves, 16 rows each
  const int lane = tid & 31;
  const int m0   = wave * 16;
  const int lrow = lane & 15;
  const int koff = (lane >> 4) * 8;      // ISA 16-bit A layout: lane[4] picks K half
  const int aRow = m0 + lrow;

  // Preload all 4 A fragments (K chunks of 32).
  v16h afr[4];
#pragma unroll
  for (int kt = 0; kt < 4; ++kt) {
    const int kb = kt * 32;
    v8h alo = *(const v8h*)&As[aRow][kb + koff];
    v8h ahi = *(const v8h*)&As[aRow][kb + 16 + koff];
    afr[kt] = __builtin_shufflevector(alo, ahi, 0,1,2,3,4,5,6,7,8,9,10,11,12,13,14,15);
  }

  v8f acc[8] = {};                       // 8 x (16x16) f32 tiles = full 128 cols
#pragma unroll
  for (int kt = 0; kt < 4; ++kt) {
    const int kb = kt * 32;
    v16h bf[8];
#pragma unroll
    for (int nt = 0; nt < 8; ++nt) {     // batch all B loads before the WMMAs
      v8h blo = *(const v8h*)&Bs[nt * 16 + lrow][kb + koff];
      v8h bhi = *(const v8h*)&Bs[nt * 16 + lrow][kb + 16 + koff];
      bf[nt] = __builtin_shufflevector(blo, bhi, 0,1,2,3,4,5,6,7,8,9,10,11,12,13,14,15);
    }
#pragma unroll
    for (int nt = 0; nt < 8; ++nt) {
      acc[nt] = __builtin_amdgcn_wmma_f32_16x16x32_f16(
          false, afr[kt], false, bf[nt], (short)0, acc[nt], false, false);
    }
  }

  // Store D tiles (C/D layout: vgpr r -> M = r + 8*lane[4], N = lane[3:0]).
#pragma unroll
  for (int nt = 0; nt < 8; ++nt) {
    int col = nt * 16 + lrow;
    float bv = bias ? bias[col] : 0.f;
#pragma unroll
    for (int r = 0; r < 8; ++r) {
      int grow = row0 + m0 + ((lane >> 4) << 3) + r;
      if (grow < M) {
        float v = acc[nt][r] + bv;
        if (do_relu) v = fmaxf(v, 0.f);
        if (Cout32) Cout32[(size_t)grow * DD + col] = v;
        if (Cout16) Cout16[(size_t)grow * DD + col] = (_Float16)v;
      }
    }
  }
}

// ---------------- attention logits per (node, head) -------------------------
__global__ void alpha_kernel(const _Float16* __restrict__ hp16,
                             const float* __restrict__ a_s,
                             const float* __restrict__ a_d,
                             float* __restrict__ as_out,
                             float* __restrict__ ad_out, int N) {
  int gid = blockIdx.x * blockDim.x + threadIdx.x;
  if (gid >= N * HH_) return;
  int n = gid >> 2, hh = gid & 3;
  const _Float16* hr = hp16 + (size_t)n * DD + hh * CC;
  const float* asr = a_s + hh * CC;
  const float* adr = a_d + hh * CC;
  float s0 = 0.f, s1 = 0.f;
#pragma unroll
  for (int c = 0; c < CC; ++c) {
    float v = (float)hr[c];
    s0 = fmaf(v, asr[c], s0);
    s1 = fmaf(v, adr[c], s1);
  }
  as_out[gid] = s0;
  ad_out[gid] = s1;
}

// -------- per-layer init: hnew = bias (scatter target), m = -inf, z = 0 -----
__global__ void layer_init_kernel(float* __restrict__ hnew,
                                  float* __restrict__ m,
                                  float* __restrict__ z,
                                  const float* __restrict__ bg, int N) {
  int gid = blockIdx.x * blockDim.x + threadIdx.x;
  if (gid >= N * DD) return;
  int d = gid & 127;
  hnew[gid] = bg[d];
  if (d < HH_) {
    int n = gid >> 7;
    m[n * HH_ + d] = -INFINITY;
    z[n * HH_ + d] = 0.f;
  }
}

// float atomic max via signed/unsigned int ordering trick (works for all finite)
__device__ __forceinline__ void atomicMaxF(float* addr, float val) {
  if (val >= 0.f) atomicMax((int*)addr, __float_as_int(val));
  else            atomicMin((unsigned int*)addr, __float_as_uint(val));
}

// -------- edge pass 1: e = leaky_relu(as[src]+ad[dst]); segment max on dst --
__global__ void edge_max_kernel(const long long* __restrict__ ei,
                                const float* __restrict__ as_,
                                const float* __restrict__ ad_,
                                float* __restrict__ m,
                                float* __restrict__ ev,
                                int E, int Etot) {
  int e = blockIdx.x * blockDim.x + threadIdx.x;
  if (e >= Etot) return;
  int s = (e < E) ? (int)ei[e]     : (e - E);   // virtual self-loops
  int d = (e < E) ? (int)ei[E + e] : (e - E);
#pragma unroll
  for (int hh = 0; hh < HH_; ++hh) {
    float v = as_[s * HH_ + hh] + ad_[d * HH_ + hh];
    v = (v > 0.f) ? v : NEG_SLOPE * v;
    ev[(size_t)e * HH_ + hh] = v;
    atomicMaxF(&m[d * HH_ + hh], v);
  }
}

// -------- edge pass 2: ev = exp(ev - m[dst]); segment sum into z ------------
__global__ void edge_exp_kernel(const long long* __restrict__ ei,
                                const float* __restrict__ m,
                                float* __restrict__ z,
                                float* __restrict__ ev,
                                int E, int Etot) {
  int e = blockIdx.x * blockDim.x + threadIdx.x;
  if (e >= Etot) return;
  int d = (e < E) ? (int)ei[E + e] : (e - E);
#pragma unroll
  for (int hh = 0; hh < HH_; ++hh) {
    float x = __expf(ev[(size_t)e * HH_ + hh] - m[d * HH_ + hh]);
    ev[(size_t)e * HH_ + hh] = x;
    atomicAdd(&z[d * HH_ + hh], x);
  }
}

// -------- aggregate: hnew[dst] += alpha * hp[src] (block per edge) ----------
__global__ void aggregate_kernel(const long long* __restrict__ ei,
                                 const _Float16* __restrict__ hp16,
                                 const float* __restrict__ ev,
                                 const float* __restrict__ z,
                                 float* __restrict__ hnew,
                                 int E) {
  int e = blockIdx.x;
  int t = threadIdx.x;                   // 0..127 channels
  int s = (e < E) ? (int)ei[e]     : (e - E);
  int d = (e < E) ? (int)ei[E + e] : (e - E);
  int hh = t >> 5;
  float alpha = ev[(size_t)e * HH_ + hh] / (z[d * HH_ + hh] + 1e-16f);
  float hv = (float)hp16[(size_t)s * DD + t];
  atomicAdd(&hnew[(size_t)d * DD + t], hv * alpha);
}

// -------- LayerNorm + ReLU + residual: h += relu(LN(hnew)); wave per node ---
__global__ void ln_relu_res_kernel(const float* __restrict__ hnew,
                                   const float* __restrict__ g,
                                   const float* __restrict__ b,
                                   float* __restrict__ h,
                                   _Float16* __restrict__ h16, int N) {
  int node = blockIdx.x * 4 + (threadIdx.x >> 5);
  if (node >= N) return;
  int lane = threadIdx.x & 31;
  const float* row = hnew + (size_t)node * DD;
  float v[4];
  float sum = 0.f;
#pragma unroll
  for (int i = 0; i < 4; ++i) { v[i] = row[lane + i * 32]; sum += v[i]; }
#pragma unroll
  for (int o = 16; o > 0; o >>= 1) sum += __shfl_xor(sum, o, 32);
  float mu = sum * (1.f / 128.f);
  float var = 0.f;
#pragma unroll
  for (int i = 0; i < 4; ++i) { float dlt = v[i] - mu; var += dlt * dlt; }
#pragma unroll
  for (int o = 16; o > 0; o >>= 1) var += __shfl_xor(var, o, 32);
  var *= (1.f / 128.f);
  float inv = rsqrtf(var + LN_EPS);
#pragma unroll
  for (int i = 0; i < 4; ++i) {
    int c = lane + i * 32;
    float y = (v[i] - mu) * inv * g[c] + b[c];
    float out = h[(size_t)node * DD + c] + fmaxf(y, 0.f);
    h[(size_t)node * DD + c] = out;
    h16[(size_t)node * DD + c] = (_Float16)out;
  }
}

// ---------------- global mean pool ------------------------------------------
__global__ void pool_zero_kernel(float* __restrict__ sums,
                                 float* __restrict__ counts, int G) {
  int gid = blockIdx.x * blockDim.x + threadIdx.x;
  if (gid >= G * DD) return;
  sums[gid] = 0.f;
  if ((gid & 127) == 0) counts[gid >> 7] = 0.f;
}

__global__ void pool_acc_kernel(const float* __restrict__ h,
                                const long long* __restrict__ batch,
                                float* __restrict__ sums,
                                float* __restrict__ counts, int N) {
  int gid = blockIdx.x * blockDim.x + threadIdx.x;
  if (gid >= N * DD) return;
  int n = gid >> 7, d = gid & 127;
  int g = (int)batch[n];
  atomicAdd(&sums[(size_t)g * DD + d], h[gid]);
  if (d == 0) atomicAdd(&counts[g], 1.f);
}

__global__ void pool_div_kernel(const float* __restrict__ sums,
                                const float* __restrict__ counts,
                                _Float16* __restrict__ pooled16, int G) {
  int gid = blockIdx.x * blockDim.x + threadIdx.x;
  if (gid >= G * DD) return;
  pooled16[gid] = (_Float16)(sums[gid] / fmaxf(counts[gid >> 7], 1.f));
}

extern "C" void kernel_launch(void* const* d_in, const int* in_sizes, int n_in,
                              void* d_out, int out_size, void* d_ws, size_t ws_size,
                              hipStream_t stream) {
  const float*     x     = (const float*)d_in[0];
  const long long* ei    = (const long long*)d_in[1];   // int64 [2,E]
  const long long* batch = (const long long*)d_in[2];   // int64 [N]
  const float* W0    = (const float*)d_in[3];
  const float* b0    = (const float*)d_in[4];
  const float* Wg    = (const float*)d_in[5];           // [L,128,128]
  const float* a_src = (const float*)d_in[6];           // [L,4,32]
  const float* a_dst = (const float*)d_in[7];
  const float* bg    = (const float*)d_in[8];           // [L,128]
  const float* ln_g  = (const float*)d_in[9];
  const float* ln_b  = (const float*)d_in[10];
  const float* W1    = (const float*)d_in[11];
  const float* b1    = (const float*)d_in[12];
  const float* W2    = (const float*)d_in[13];
  const float* b2    = (const float*)d_in[14];

  const int N    = in_sizes[0] / 9;
  const int E    = in_sizes[1] / 2;
  const int G    = out_size / DD;
  const int Etot = E + N;
  const int L    = 4;

  // f32 region then f16 region (all sub-offsets stay 16B aligned)
  float* ws = (float*)d_ws;
  size_t off = 0;
  float* h      = ws + off; off += (size_t)N * DD;
  float* hnew   = ws + off; off += (size_t)N * DD;
  float* as_buf = ws + off; off += (size_t)N * HH_;
  float* ad_buf = ws + off; off += (size_t)N * HH_;
  float* mbuf   = ws + off; off += (size_t)N * HH_;
  float* zbuf   = ws + off; off += (size_t)N * HH_;
  float* ev     = ws + off; off += (size_t)Etot * HH_;
  float* sums   = ws + off; off += (size_t)G * DD;
  float* counts = ws + off; off += (size_t)G;
  _Float16* ws16 = (_Float16*)(ws + off);
  size_t off16 = 0;
  _Float16* h16      = ws16 + off16; off16 += (size_t)N * DD;
  _Float16* hp16     = ws16 + off16; off16 += (size_t)N * DD;
  _Float16* pooled16 = ws16 + off16; off16 += (size_t)G * DD;
  _Float16* hidden16 = ws16 + off16; off16 += (size_t)G * DD;

  const int ND = N * DD;
  dim3 blk256(256);

  node_embed_kernel<<<(ND + 255) / 256, blk256, 0, stream>>>(x, W0, b0, h, h16, N);

  for (int l = 0; l < L; ++l) {
    // projection straight to f16 (feeds alpha + gather); no f32 copy needed
    gemm128_wmma_kernel<<<(N + 127) / 128, 256, 0, stream>>>(
        h16, Wg + (size_t)l * DD * DD, nullptr, nullptr, hp16, N, 0);
    alpha_kernel<<<(N * HH_ + 255) / 256, blk256, 0, stream>>>(
        hp16, a_src + l * HH_ * CC, a_dst + l * HH_ * CC, as_buf, ad_buf, N);
    layer_init_kernel<<<(ND + 255) / 256, blk256, 0, stream>>>(
        hnew, mbuf, zbuf, bg + l * DD, N);
    edge_max_kernel<<<(Etot + 255) / 256, blk256, 0, stream>>>(
        ei, as_buf, ad_buf, mbuf, ev, E, Etot);
    edge_exp_kernel<<<(Etot + 255) / 256, blk256, 0, stream>>>(
        ei, mbuf, zbuf, ev, E, Etot);
    aggregate_kernel<<<Etot, 128, 0, stream>>>(ei, hp16, ev, zbuf, hnew, E);
    ln_relu_res_kernel<<<(N + 3) / 4, 128, 0, stream>>>(
        hnew, ln_g + l * DD, ln_b + l * DD, h, h16, N);
  }

  pool_zero_kernel<<<(G * DD + 255) / 256, blk256, 0, stream>>>(sums, counts, G);
  pool_acc_kernel<<<(ND + 255) / 256, blk256, 0, stream>>>(h, batch, sums, counts, N);
  pool_div_kernel<<<(G * DD + 255) / 256, blk256, 0, stream>>>(sums, counts, pooled16, G);

  // output MLP: relu(pooled@W1+b1) (f16 hidden) @ W2 + b2 -> f32 d_out
  gemm128_wmma_kernel<<<(G + 127) / 128, 256, 0, stream>>>(
      pooled16, W1, b1, nullptr, hidden16, G, 1);
  gemm128_wmma_kernel<<<(G + 127) / 128, 256, 0, stream>>>(
      hidden16, W2, b2, (float*)d_out, nullptr, G, 0);
}
